// SlotAttention_26336739459479
// MI455X (gfx1250) — matrix operand order repaired and checked by reading, not compiled, not run on hardware
//
#include <hip/hip_runtime.h>

// ---------------- sizes ----------------
#define LL   8192
#define BB   16
#define DD   256
#define SS   10
#define HID  1024
#define ITERS 3
#define NEGF (-3.402823466e38f)

// ---------------- CDNA5 WMMA types ----------------
typedef __attribute__((ext_vector_type(16))) __bf16 v16bf;
typedef __attribute__((ext_vector_type(8)))  float  v8f;
typedef __attribute__((ext_vector_type(4)))  unsigned int v4u;

union ABu { v16bf v; v4u q[2]; unsigned int u[8]; };

static __device__ __forceinline__ unsigned short f2bf(float f) {
    unsigned int u = __builtin_bit_cast(unsigned int, f);
    u = (u + 0x7FFFu + ((u >> 16) & 1u)) >> 16;   // RNE (NaN ignored)
    return (unsigned short)u;
}
static __device__ __forceinline__ float bf2f(unsigned short h) {
    return __builtin_bit_cast(float, (unsigned int)h << 16);
}

// =====================================================================
// Kernel 0: swizzle Wk / Wv (f32, 256x256 row-major KxN) into bf16
// B-fragment order with each lane's 8 dwords CONTIGUOUS:
//   kvS[mat*32768 + ((nt*8+ks)*32 + lane)*8 + i] = pack(W[k0,n], W[k0+1,n])
//   n = nt*16 + (lane&15), k0 = ks*32 + (lane>>4)*16 + 2*i
// B fragment load is then 2 x global_load_b128 per lane off a single
// per-wave base pointer with compile-time immediate offsets.
// =====================================================================
__global__ __launch_bounds__(256) void k_swz(const float* __restrict__ Wk,
                                             const float* __restrict__ Wv,
                                             unsigned int* __restrict__ kvS) {
    int tid = blockIdx.x * 256 + threadIdx.x;        // 65536 threads
    int mat = tid >> 15;
    int r   = tid & 32767;
    int i    = r & 7;
    int lane = (r >> 3) & 31;
    int ks   = (r >> 8) & 7;
    int nt   = (r >> 11) & 15;
    int n  = nt * 16 + (lane & 15);
    int k0 = ks * 32 + (lane >> 4) * 16 + 2 * i;
    const float* W = mat ? Wv : Wk;
    unsigned int lo = f2bf(W[(size_t)k0 * DD + n]);
    unsigned int hi = f2bf(W[(size_t)(k0 + 1) * DD + n]);
    kvS[(size_t)mat * 32768 + r] = lo | (hi << 16);
}

// =====================================================================
// Kernel 1: fused input layer-norm + K/V GEMM (bf16 WMMA, f32 accum).
// Block = 256 threads (8 waves), M-tile = 64 rows, N = 256 for K and V.
// A fragments hoisted (loaded once per wave, 2 x ds_load_b128 each);
// B fragments: single base pointer + immediate offsets;
// outputs repacked through wave-private LDS -> global_store_b128.
// =====================================================================
#define WPITCH 132   // dwords per LDS row (264 bf16; 4-bank stride -> conflict-free)
__global__ __launch_bounds__(256) void k_ln_kv(const float* __restrict__ inp,
                                               const float* __restrict__ lnw,
                                               const float* __restrict__ lnb,
                                               const unsigned int* __restrict__ kvS,
                                               const float* __restrict__ bk,
                                               const float* __restrict__ bv,
                                               unsigned short* __restrict__ Kb,
                                               unsigned short* __restrict__ Vb) {
    __shared__ unsigned int xt[64 * WPITCH];
    __shared__ unsigned int ot[8][16 * 32];          // per-wave 16x64 bf16 repack tile
    const int tid = threadIdx.x, lane = tid & 31, w = tid >> 5;
    const int bid = blockIdx.x;

    // ---- phase 1: layer-norm 64 rows -> bf16 LDS tile ----
    unsigned short* xts = (unsigned short*)xt;
    for (int rr = w * 8; rr < w * 8 + 8; ++rr) {
        int gr = bid * 64 + rr;                 // row of x = (b, l)
        int b = gr >> 13, l = gr & (LL - 1);
        const float* src = inp + ((size_t)l * BB + b) * DD;   // inputs is (L,B,D)
        float x[8]; float s = 0.f;
        #pragma unroll
        for (int t = 0; t < 8; ++t) { x[t] = src[t * 32 + lane]; s += x[t]; }
        #pragma unroll
        for (int m = 16; m >= 1; m >>= 1) s += __shfl_xor(s, m, 32);
        float mean = s * (1.0f / 256.0f);
        float vs = 0.f;
        #pragma unroll
        for (int t = 0; t < 8; ++t) { float d = x[t] - mean; vs += d * d; }
        #pragma unroll
        for (int m = 16; m >= 1; m >>= 1) vs += __shfl_xor(vs, m, 32);
        float rstd = rsqrtf(vs * (1.0f / 256.0f) + 1e-5f);
        #pragma unroll
        for (int t = 0; t < 8; ++t) {
            int d = t * 32 + lane;
            float y = (x[t] - mean) * rstd * lnw[d] + lnb[d];
            xts[rr * 264 + d] = f2bf(y);
        }
    }
    __syncthreads();

    // ---- phase 2: WMMA; wave w -> row-tile (w&3), N-half (w>>2) ----
    const int rt = w & 3, nh = w >> 2;
    const int rb = rt * 16;
    const int m  = lane & 15, hf = lane >> 4;

    // hoist all 8 A fragments for this wave's 16 rows
    ABu afr[8];
    const unsigned int* xr = xt + (rb + m) * WPITCH;
    #pragma unroll
    for (int ks = 0; ks < 8; ++ks) {
        int kb2 = ks * 16 + hf * 8;              // dword offset of K-pairs
        afr[ks].q[0] = *(const v4u*)(xr + kb2);
        afr[ks].q[1] = *(const v4u*)(xr + kb2 + 4);
    }

    // single runtime base; everything else is an immediate offset
    const unsigned int* pbase = kvS + (size_t)nh * 16384 + (size_t)lane * 8;

    #pragma unroll
    for (int mat = 0; mat < 2; ++mat) {
        const float* bias = mat ? bv : bk;
        unsigned short* dst = mat ? Vb : Kb;
        #pragma unroll
        for (int g = 0; g < 2; ++g) {
            v8f acc[4];
            #pragma unroll
            for (int j = 0; j < 4; ++j) acc[j] = (v8f){};
            #pragma unroll
            for (int ks = 0; ks < 8; ++ks) {
                __builtin_prefetch(pbase + mat * 32768 + g * 8192 + ks * 256 + 2048, 0, 1);
                #pragma unroll
                for (int j = 0; j < 4; ++j) {
                    const unsigned int* p =
                        pbase + mat * 32768 + (g * 4 + j) * 2048 + ks * 256;
                    ABu bfr;
                    bfr.q[0] = *(const v4u*)(p);
                    bfr.q[1] = *(const v4u*)(p + 4);
                    acc[j] = __builtin_amdgcn_wmma_f32_16x16x32_bf16(
                        false, afr[ks].v, false, bfr.v, (short)0, acc[j], false, false);
                }
            }
            // ---- epilogue: bias, bf16-pack into wave LDS tile, b128 stores ----
            unsigned short* os = (unsigned short*)ot[w];
            #pragma unroll
            for (int j = 0; j < 4; ++j) {
                float bsv = bias[nh * 128 + g * 64 + j * 16 + m];
                #pragma unroll
                for (int r = 0; r < 8; ++r)
                    os[(r + 8 * hf) * 64 + j * 16 + m] = f2bf(acc[j][r] + bsv);
            }
            // wave-private: DS in-order per wave, no barrier needed
            #pragma unroll
            for (int c = 0; c < 4; ++c) {
                int chunk = lane + 32 * c;        // 0..127
                int row = chunk >> 3;             // 0..15
                int off = (chunk & 7) * 4;        // dword offset within 32-dword row
                v4u val = *(const v4u*)&ot[w][row * 32 + off];
                size_t grow = (size_t)bid * 64 + rb + row;
                unsigned int* gdst = (unsigned int*)(dst + grow * DD) + nh * 64 + g * 32 + off;
                *(v4u*)gdst = val;
            }
        }
    }
}

// =====================================================================
// Kernel 2 (per iteration): zero accumulators, s = LN(slots), q = s@Wq+bq
// Stored bf16, padded to 16 slots (rows 10..15 = 0). One block per b.
// =====================================================================
__global__ __launch_bounds__(256) void k_q(const float* __restrict__ slots,
                                           const float* __restrict__ lnw,
                                           const float* __restrict__ lnb,
                                           const float* __restrict__ Wq,
                                           const float* __restrict__ bq,
                                           unsigned short* __restrict__ Qb,
                                           float* __restrict__ numer,
                                           float* __restrict__ denom) {
    __shared__ float sln[SS * DD];
    const int b = blockIdx.x, tid = threadIdx.x, lane = tid & 31, w = tid >> 5;
    for (int it = 0; it < 16; ++it)
        numer[(size_t)b * 16 * DD + it * DD + tid] = 0.f;
    if (tid < 16) denom[b * 16 + tid] = 0.f;
    for (int r = w; r < SS; r += 8) {
        const float* src = slots + ((size_t)b * SS + r) * DD;
        float x[8]; float s = 0.f;
        #pragma unroll
        for (int t = 0; t < 8; ++t) { x[t] = src[t * 32 + lane]; s += x[t]; }
        #pragma unroll
        for (int m = 16; m >= 1; m >>= 1) s += __shfl_xor(s, m, 32);
        float mean = s * (1.0f / 256.0f);
        float vs = 0.f;
        #pragma unroll
        for (int t = 0; t < 8; ++t) { float d = x[t] - mean; vs += d * d; }
        #pragma unroll
        for (int m = 16; m >= 1; m >>= 1) vs += __shfl_xor(vs, m, 32);
        float rstd = rsqrtf(vs * (1.0f / 256.0f) + 1e-5f);
        #pragma unroll
        for (int t = 0; t < 8; ++t) {
            int d = t * 32 + lane;
            sln[r * DD + d] = (x[t] - mean) * rstd * lnw[d] + lnb[d];
        }
    }
    __syncthreads();
    for (int i = 0; i < SS; ++i) {
        float s = bq[tid];
        for (int d = 0; d < DD; ++d) s += sln[i * DD + d] * Wq[(size_t)d * DD + tid];
        Qb[((size_t)b * 16 + i) * DD + tid] = f2bf(s);
    }
    for (int i = SS; i < 16; ++i) Qb[((size_t)b * 16 + i) * DD + tid] = 0;
}

// =====================================================================
// Kernel 3 (per iteration): dots via WMMA, per-j softmax over slots,
// accumulate numer[i,d] and denom[i] with atomics. Grid = (B, L/JCH).
// =====================================================================
#define JCH 512
__global__ __launch_bounds__(256) void k_attn(const unsigned short* __restrict__ Kb,
                                              const unsigned short* __restrict__ Vb,
                                              const unsigned short* __restrict__ Qb,
                                              const int* __restrict__ mask,
                                              float* __restrict__ numer,
                                              float* __restrict__ denom) {
    __shared__ float wl[JCH * 16];
    __shared__ float dl[16];
    const int tid = threadIdx.x, lane = tid & 31, w = tid >> 5;
    const int b = blockIdx.x, jbase = blockIdx.y * JCH;
    if (tid < 16) dl[tid] = 0.f;
    __syncthreads();

    const int m = lane & 15, hf = lane >> 4;
    const int nvalid = hf ? 2 : 8;
    const unsigned int* qd = (const unsigned int*)(Qb + (size_t)b * 16 * DD);
    const unsigned int* kd = (const unsigned int*)(Kb + (size_t)b * LL * DD);
    float dacc[8] = {};

    // hoist q fragments (invariant over j-tiles): 2 x b128 per k-step
    ABu afr[8];
    #pragma unroll
    for (int ks = 0; ks < 8; ++ks) {
        int ka = ks * 16 + hf * 8;
        afr[ks].q[0] = *(const v4u*)(qd + m * 128 + ka);
        afr[ks].q[1] = *(const v4u*)(qd + m * 128 + ka + 4);
    }

    for (int tt = 0; tt < JCH / 16 / 8; ++tt) {    // 4 j-tiles per wave
        int j0 = jbase + (w + 8 * tt) * 16;
        int j  = j0 + m;
        v8f acc = (v8f){};
        const unsigned int* kr = kd + (size_t)j * (DD / 2);
        __builtin_prefetch(kr + 1024, 0, 1);
        #pragma unroll
        for (int ks = 0; ks < 8; ++ks) {
            int kb = ks * 16 + hf * 8;
            ABu bfr;
            bfr.q[0] = *(const v4u*)(kr + kb);
            bfr.q[1] = *(const v4u*)(kr + kb + 4);
            acc = __builtin_amdgcn_wmma_f32_16x16x32_bf16(
                false, afr[ks].v, false, bfr.v, (short)0, acc, false, false);
        }
        bool msk = mask[(size_t)b * LL + j] != 0;
        float vv[8];
        #pragma unroll
        for (int r = 0; r < 8; ++r) vv[r] = msk ? NEGF : acc[r] * 0.0625f;
        float mymax = NEGF;
        #pragma unroll
        for (int r = 0; r < 8; ++r) if (r < nvalid) mymax = fmaxf(mymax, vv[r]);
        float mx = fmaxf(mymax, __shfl_xor(mymax, 16, 32));
        float e[8]; float es = 0.f;
        #pragma unroll
        for (int r = 0; r < 8; ++r) { e[r] = (r < nvalid) ? __expf(vv[r] - mx) : 0.f; es += e[r]; }
        float tot = es + __shfl_xor(es, 16, 32);
        float inv = 1.0f / tot;
        #pragma unroll
        for (int r = 0; r < 8; ++r) {
            float wgt = e[r] * inv;
            dacc[r] += wgt;
            wl[(j0 - jbase + m) * 16 + r + 8 * hf] = wgt;
        }
    }
    #pragma unroll
    for (int r = 0; r < 8; ++r) {
        float v = dacc[r];
        #pragma unroll
        for (int mm = 1; mm <= 8; mm <<= 1) v += __shfl_xor(v, mm, 32);
        dacc[r] = v;
    }
    if (m == 0) {
        #pragma unroll
        for (int r = 0; r < 8; ++r)
            if (r < nvalid) atomicAdd(&dl[r + 8 * hf], dacc[r]);
    }
    __syncthreads();
    if (tid < 16) atomicAdd(&denom[b * 16 + tid], dl[tid]);

    // numer: thread -> (d pair, j half); one b32 load covers 2 d's
    const int th = tid >> 7;                 // j-half
    const int dp = (tid & 127) * 2;          // d pair base
    float a0[SS] = {}, a1[SS] = {};
    const unsigned int* vbase =
        (const unsigned int*)(Vb + ((size_t)b * LL + jbase) * DD) + (dp >> 1);
    const int jlo = th * (JCH / 2), jhi = jlo + JCH / 2;
    for (int jj = jlo; jj < jhi; ++jj) {
        __builtin_prefetch(vbase + (size_t)(jj + 8) * 128, 0, 1);
        unsigned int pv = vbase[(size_t)jj * 128];
        float v0 = bf2f((unsigned short)pv);
        float v1 = bf2f((unsigned short)(pv >> 16));
        #pragma unroll
        for (int i = 0; i < SS; ++i) {
            float wv = wl[jj * 16 + i];
            a0[i] += wv * v0;
            a1[i] += wv * v1;
        }
    }
    #pragma unroll
    for (int i = 0; i < SS; ++i) {
        atomicAdd(&numer[((size_t)b * 16 + i) * DD + dp],     a0[i]);
        atomicAdd(&numer[((size_t)b * 16 + i) * DD + dp + 1], a1[i]);
    }
}

// =====================================================================
// Kernel 4 (per iteration): slots += numer/denom; then MLP residual.
// One block per (b,s) row, 256 threads.
// =====================================================================
__global__ __launch_bounds__(256) void k_upd(float* __restrict__ slots,
                                             const float* __restrict__ numer,
                                             const float* __restrict__ denom,
                                             const float* __restrict__ lnw,
                                             const float* __restrict__ lnb,
                                             const float* __restrict__ W1, const float* __restrict__ b1,
                                             const float* __restrict__ W2, const float* __restrict__ b2,
                                             const float* __restrict__ W3, const float* __restrict__ b3) {
    __shared__ float xln[DD], h1[HID], h2[HID], red[DD];
    const int bs = blockIdx.x, b = bs / SS, s = bs % SS, t = threadIdx.x;
    float den = denom[b * 16 + s] + 1e-8f;
    float xs = slots[((size_t)b * SS + s) * DD + t]
             + numer[((size_t)b * 16 + s) * DD + t] / den;
    red[t] = xs; __syncthreads();
    for (int st = 128; st >= 1; st >>= 1) { if (t < st) red[t] += red[t + st]; __syncthreads(); }
    float mean = red[0] * (1.0f / 256.0f); __syncthreads();
    float dx = xs - mean;
    red[t] = dx * dx; __syncthreads();
    for (int st = 128; st >= 1; st >>= 1) { if (t < st) red[t] += red[t + st]; __syncthreads(); }
    float var = red[0] * (1.0f / 256.0f); __syncthreads();
    xln[t] = dx * rsqrtf(var + 1e-5f) * lnw[t] + lnb[t];
    __syncthreads();
    for (int o = t; o < HID; o += 256) {
        float a = b1[o];
        for (int d = 0; d < DD; ++d) a += xln[d] * W1[(size_t)d * HID + o];
        h1[o] = fmaxf(a, 0.f);
    }
    __syncthreads();
    for (int o = t; o < HID; o += 256) {
        float a = b2[o];
        for (int d = 0; d < HID; ++d) a += h1[d] * W2[(size_t)d * HID + o];
        h2[o] = fmaxf(a, 0.f);
    }
    __syncthreads();
    float a = b3[t];
    for (int d = 0; d < HID; ++d) a += h2[d] * W3[(size_t)d * DD + t];
    slots[((size_t)b * SS + s) * DD + t] = xs + a;
}

// =====================================================================
// Kernel 5: final layer-norm -> d_out (f32, B x S x D)
// =====================================================================
__global__ __launch_bounds__(256) void k_final(const float* __restrict__ slots,
                                               const float* __restrict__ lnw,
                                               const float* __restrict__ lnb,
                                               float* __restrict__ out) {
    __shared__ float red[DD];
    const int bs = blockIdx.x, t = threadIdx.x;
    float x = slots[(size_t)bs * DD + t];
    red[t] = x; __syncthreads();
    for (int st = 128; st >= 1; st >>= 1) { if (t < st) red[t] += red[t + st]; __syncthreads(); }
    float mean = red[0] * (1.0f / 256.0f); __syncthreads();
    float dx = x - mean;
    red[t] = dx * dx; __syncthreads();
    for (int st = 128; st >= 1; st >>= 1) { if (t < st) red[t] += red[t + st]; __syncthreads(); }
    float var = red[0] * (1.0f / 256.0f);
    out[(size_t)bs * DD + t] = dx * rsqrtf(var + 1e-5f) * lnw[t] + lnb[t];
}

__global__ __launch_bounds__(256) void k_init(const float* __restrict__ src,
                                              float* __restrict__ dst) {
    int i = blockIdx.x * 256 + threadIdx.x;
    dst[i] = src[i];
}

// =====================================================================
// host-side orchestration
// =====================================================================
extern "C" void kernel_launch(void* const* d_in, const int* in_sizes, int n_in,
                              void* d_out, int out_size, void* d_ws, size_t ws_size,
                              hipStream_t stream) {
    const float* inputs  = (const float*)d_in[0];
    const int*   mask    = (const int*)  d_in[1];
    const float* slots0  = (const float*)d_in[2];
    const float* ln_in_w = (const float*)d_in[3];
    const float* ln_in_b = (const float*)d_in[4];
    const float* ln_sl_w = (const float*)d_in[5];
    const float* ln_sl_b = (const float*)d_in[6];
    const float* ln_mlp_w= (const float*)d_in[7];
    const float* ln_mlp_b= (const float*)d_in[8];
    const float* ln_out_w= (const float*)d_in[9];
    const float* ln_out_b= (const float*)d_in[10];
    const float* Wq = (const float*)d_in[11];
    const float* bq = (const float*)d_in[12];
    const float* Wk = (const float*)d_in[13];
    const float* bk = (const float*)d_in[14];
    const float* Wv = (const float*)d_in[15];
    const float* bv = (const float*)d_in[16];
    const float* W1 = (const float*)d_in[17];
    const float* b1 = (const float*)d_in[18];
    const float* W2 = (const float*)d_in[19];
    const float* b2 = (const float*)d_in[20];
    const float* W3 = (const float*)d_in[21];
    const float* b3 = (const float*)d_in[22];

    char* ws = (char*)d_ws;
    size_t off = 0;
    unsigned short* Kb = (unsigned short*)(ws + off); off += (size_t)BB * LL * DD * 2;
    unsigned short* Vb = (unsigned short*)(ws + off); off += (size_t)BB * LL * DD * 2;
    unsigned int* kvS  = (unsigned int*)(ws + off);   off += 2u * 32768u * 4;  // Wk|Wv swizzled
    unsigned short* Qb = (unsigned short*)(ws + off); off += (size_t)BB * 16 * DD * 2;
    float* numer = (float*)(ws + off);                off += (size_t)BB * 16 * DD * 4;
    float* denom = (float*)(ws + off);                off += (size_t)BB * 16 * 4;
    float* slotsW = (float*)(ws + off);               off += (size_t)BB * SS * DD * 4;
    (void)ws_size; (void)n_in; (void)in_sizes; (void)out_size;

    k_swz<<<256, 256, 0, stream>>>(Wk, Wv, kvS);
    k_init<<<(BB * SS * DD) / 256, 256, 0, stream>>>(slots0, slotsW);
    k_ln_kv<<<(BB * LL) / 64, 256, 0, stream>>>(inputs, ln_in_w, ln_in_b,
                                                kvS, bk, bv, Kb, Vb);
    for (int it = 0; it < ITERS; ++it) {
        k_q<<<BB, 256, 0, stream>>>(slotsW, ln_sl_w, ln_sl_b, Wq, bq, Qb, numer, denom);
        k_attn<<<dim3(BB, LL / JCH), 256, 0, stream>>>(Kb, Vb, Qb, mask, numer, denom);
        k_upd<<<BB * SS, 256, 0, stream>>>(slotsW, numer, denom, ln_mlp_w, ln_mlp_b,
                                           W1, b1, W2, b2, W3, b3);
    }
    k_final<<<BB * SS, 256, 0, stream>>>(slotsW, ln_out_w, ln_out_b, (float*)d_out);
}
